// DeformTransWorldFeat_35734127902969
// MI455X (gfx1250) — matrix operand order repaired and checked by reading, not compiled, not run on hardware
//
#include <hip/hip_runtime.h>

// ---------------- constants ----------------
constexpr int kC   = 128;
constexpr int kH   = 40;
constexpr int kW   = 120;
constexpr int kRH  = 80;
constexpr int kRW  = 240;
constexpr int kHW  = kH * kW;        // 4800
constexpr int kNL  = 4;
constexpr int kLq  = kNL * kHW;      // 19200
constexpr int kNH  = 8;
constexpr int kNP  = 4;
constexpr int kDFF = 512;
constexpr int kL   = 3;

typedef __bf16 bf16_t;
typedef __bf16 bf16x8  __attribute__((ext_vector_type(8)));
typedef __bf16 bf16x16 __attribute__((ext_vector_type(16)));
typedef float  floatx8 __attribute__((ext_vector_type(8)));

__device__ __forceinline__ bf16_t f2b(float f) {
  unsigned int u = __builtin_bit_cast(unsigned int, f);
  unsigned int r = (u + 0x7FFFu + ((u >> 16) & 1u)) >> 16;
  unsigned short h = (unsigned short)r;
  return __builtin_bit_cast(bf16_t, h);
}

// ---------------- fp32 -> bf16 convert ----------------
__global__ void cvt_bf16_kernel(const float* __restrict__ s, bf16_t* __restrict__ d, int n) {
  int t = blockIdx.x * blockDim.x + threadIdx.x;
  if (t < n) d[t] = f2b(s[t]);
}

// ---------------- sine positional embedding + level embedding ----------------
__global__ void pos_kernel(const float* __restrict__ lvl_emb, float* __restrict__ pos) {
  int t = blockIdx.x * blockDim.x + threadIdx.x;
  if (t >= kLq * kC) return;
  int c = t & 127;
  int q = t >> 7;
  int n = q / kHW;
  int rem = q % kHW;
  int y = rem / kW;
  int x = rem % kW;
  const float scale = 6.28318530717958647692f;
  float e = (c < 64) ? ((float)(y + 1) / (40.0f + 1e-6f)) * scale
                     : ((float)(x + 1) / (120.0f + 1e-6f)) * scale;
  int ci = c & 63;
  float dimt = powf(10000.0f, (2.0f * (float)(ci >> 1)) / 64.0f);
  float p = e / dimt;
  float v = (ci & 1) ? cosf(p) : sinf(p);
  pos[t] = v + lvl_emb[n * kC + c];
}

// ---------------- stride-2 3x3 conv + ReLU, NCHW in -> token-major (n,y,x,c) out ----------------
__global__ void dconv_kernel(const float* __restrict__ x, const float* __restrict__ w,
                             const float* __restrict__ b, float* __restrict__ src) {
  __shared__ float patch[9 * kC];
  int q = blockIdx.x;                  // output pixel over all levels
  int n = q / kHW;
  int rem = q % kHW;
  int oy = rem / kW;
  int ox = rem % kW;
  int iy0 = oy * 2 - 1;
  int ix0 = ox * 2 - 1;
  int t = threadIdx.x;
  for (int e = t; e < 9 * kC; e += 128) {
    int tap = e >> 7;
    int ci  = e & 127;
    int ky = tap / 3, kx = tap % 3;
    int iy = iy0 + ky, ix = ix0 + kx;
    float v = 0.0f;
    if (iy >= 0 && iy < kRH && ix >= 0 && ix < kRW)
      v = x[((size_t)(n * kC + ci) * kRH + iy) * kRW + ix];
    patch[e] = v;
  }
  __syncthreads();
  int o = t;
  float acc = b[o];
  const float* wo = w + (size_t)o * 1152;   // (O,I,3,3)
  for (int ci = 0; ci < kC; ++ci) {
    float* pp = &patch[ci];
    const float* wp = wo + ci * 9;
#pragma unroll
    for (int tap = 0; tap < 9; ++tap)
      acc = fmaf(pp[tap * kC], wp[tap], acc);
  }
  src[(size_t)q * kC + o] = fmaxf(acc, 0.0f);
}

// ---------------- pack src (bf16) and q = src + pos (bf16) ----------------
__global__ void pack_q_kernel(const float* __restrict__ src, const float* __restrict__ pos,
                              bf16_t* __restrict__ sb, bf16_t* __restrict__ qb) {
  int t = blockIdx.x * blockDim.x + threadIdx.x;
  if (t >= kLq * kC) return;
  float s = src[t];
  sb[t] = f2b(s);
  qb[t] = f2b(s + pos[t]);
}

// ---------------- WMMA bf16 GEMM: C[M,N] = A[M,K] * W[N,K]^T + bias ----------------
// A fragment (16x32): lane m%16 holds A row, K sequential; half-lanes split K 0-15 / 16-31.
__device__ __forceinline__ bf16x16 load_frag_a(const bf16_t* __restrict__ p) {
  bf16x8 lo = *(const bf16x8*)(p);
  bf16x8 hi = *(const bf16x8*)(p + 8);
  return __builtin_shufflevector(lo, hi, 0, 1, 2, 3, 4, 5, 6, 7, 8, 9, 10, 11, 12, 13, 14, 15);
}
// B fragment (32x16): lane n%16 holds W row, K sequential; half-lanes split K 0-15 / 16-31.
__device__ __forceinline__ bf16x16 load_frag_b(const bf16_t* __restrict__ p) {
  bf16x8 lo = *(const bf16x8*)(p);
  bf16x8 hi = *(const bf16x8*)(p + 8);
  return __builtin_shufflevector(lo, hi, 0, 1, 2, 3, 4, 5, 6, 7, 8, 9, 10, 11, 12, 13, 14, 15);
}

// EPI: 0 = +bias -> f32 ; 1 = +bias,ReLU -> bf16 ; 2 = +bias,ReLU -> f32
// Wave tile 32x64 (2 A-frags x 4 B-frags -> 8 WMMA per K-step); block = 4 waves
// arranged 2 row-strips x 2 col-strips -> 64x128 block tile. No predication:
// EXEC stays all-ones through every WMMA (M % 64 == 0, N % 128 == 0).
template <int EPI>
__global__ void wmma_gemm_kernel(const bf16_t* __restrict__ A, const bf16_t* __restrict__ Bw,
                                 const float* __restrict__ bias,
                                 float* __restrict__ Cf, bf16_t* __restrict__ Cb,
                                 int M, int N, int K) {
  const int lane = threadIdx.x & 31;
  const int wave = threadIdx.x >> 5;
  const int r    = lane & 15;
  const int hi   = lane >> 4;
  const int m0   = blockIdx.x * 64  + (wave & 1) * 32;   // 32-row strip
  const int n0   = blockIdx.y * 128 + (wave >> 1) * 64;  // 64-col strip

  floatx8 acc[2][4] = {};

  const bf16_t* aptr0 = A + (size_t)(m0 + r) * K + hi * 16;
  const bf16_t* aptr1 = aptr0 + (size_t)16 * K;
  const bf16_t* bptr  = Bw + (size_t)(n0 + r) * K + hi * 16;

  for (int k0 = 0; k0 < K; k0 += 32) {
    bf16x16 a0 = load_frag_a(aptr0 + k0);
    bf16x16 a1 = load_frag_a(aptr1 + k0);
    if (k0 + 32 < K) {
      __builtin_prefetch(aptr0 + k0 + 32, 0, 3);  // global_prefetch_b8, near-cache
      __builtin_prefetch(aptr1 + k0 + 32, 0, 3);
    }
#pragma unroll
    for (int j = 0; j < 4; ++j) {
      bf16x16 b = load_frag_b(bptr + (size_t)(j * 16) * K + k0);
      acc[0][j] = __builtin_amdgcn_wmma_f32_16x16x32_bf16(
          false, a0, false, b, (short)0, acc[0][j], false, false);
      acc[1][j] = __builtin_amdgcn_wmma_f32_16x16x32_bf16(
          false, a1, false, b, (short)0, acc[1][j], false, false);
    }
  }

#pragma unroll
  for (int i = 0; i < 2; ++i) {
#pragma unroll
    for (int j = 0; j < 4; ++j) {
      int col = n0 + j * 16 + r;
      float bb = bias ? bias[col] : 0.0f;
#pragma unroll
      for (int v = 0; v < 8; ++v) {
        int row = m0 + i * 16 + hi * 8 + v;   // D layout: half-lane selects row octet
        float o = acc[i][j][v] + bb;
        if (EPI >= 1) o = fmaxf(o, 0.0f);
        if (EPI == 1) Cb[(size_t)row * N + col] = f2b(o);
        else          Cf[(size_t)row * N + col] = o;
      }
    }
  }
}

// ---------------- softmax over 16 contiguous values per (q, head) ----------------
__global__ void softmax16_kernel(float* __restrict__ a) {
  int t = blockIdx.x * blockDim.x + threadIdx.x;
  if (t >= kLq * kNH) return;
  float* p = a + (size_t)t * 16;
  float mx = p[0];
#pragma unroll
  for (int k = 1; k < 16; ++k) mx = fmaxf(mx, p[k]);
  float s = 0.0f;
#pragma unroll
  for (int k = 0; k < 16; ++k) { float e = expf(p[k] - mx); p[k] = e; s += e; }
  float inv = 1.0f / s;
#pragma unroll
  for (int k = 0; k < 16; ++k) p[k] *= inv;
}

// ---------------- deformable attention bilinear sampling ----------------
// block = one query (128 threads: h = t>>4, d = t&15); output bf16 [q, h*16+d]
__global__ void deform_kernel(const float* __restrict__ val, const float* __restrict__ off,
                              const float* __restrict__ attn, bf16_t* __restrict__ outb) {
  int q = blockIdx.x;
  int t = threadIdx.x;
  int h = t >> 4;
  int rem = q % kHW;
  int qy = rem / kW;
  int qx = rem % kW;
  float rx = ((float)qx + 0.5f) / 120.0f;
  float ry = ((float)qy + 0.5f) / 40.0f;
  const float* offq = off + (size_t)q * 256;
  const float* attq = attn + (size_t)q * 128;
  float acc = 0.0f;
  for (int l = 0; l < kNL; ++l) {
    const float* vbase = val + ((size_t)l * kHW) * kC + t;   // channel h*16+d == t
#pragma unroll
    for (int p = 0; p < kNP; ++p) {
      int oi = ((h * kNL + l) * kNP + p) * 2;
      float lx = rx + offq[oi]     * (1.0f / 120.0f);
      float ly = ry + offq[oi + 1] * (1.0f / 40.0f);
      float aw = attq[h * 16 + l * kNP + p];
      float sx = lx * 120.0f - 0.5f;
      float sy = ly * 40.0f  - 0.5f;
      float x0f = floorf(sx), y0f = floorf(sy);
      int x0 = (int)x0f, y0 = (int)y0f;
      float fx = sx - x0f, fy = sy - y0f;
#pragma unroll
      for (int dy = 0; dy < 2; ++dy) {
#pragma unroll
        for (int dx = 0; dx < 2; ++dx) {
          int xi = x0 + dx, yi = y0 + dy;
          float wgt = (dx ? fx : 1.0f - fx) * (dy ? fy : 1.0f - fy);
          if (xi >= 0 && xi < kW && yi >= 0 && yi < kH)
            acc += aw * wgt * vbase[(size_t)(yi * kW + xi) * kC];
        }
      }
    }
  }
  outb[(size_t)q * kC + t] = f2b(acc);
}

// ---------------- residual + LayerNorm; emit fp32 src, bf16 src, bf16 q ----------------
__global__ void resln_kernel(float* __restrict__ src, const float* __restrict__ y,
                             const float* __restrict__ g, const float* __restrict__ b,
                             const float* __restrict__ pos,
                             bf16_t* __restrict__ sb, bf16_t* __restrict__ qb) {
  __shared__ float sh[kC];
  int q = blockIdx.x;
  int c = threadIdx.x;
  size_t idx = (size_t)q * kC + c;
  float v = src[idx] + y[idx];
  sh[c] = v;
  __syncthreads();
  for (int s = 64; s > 0; s >>= 1) {
    if (c < s) sh[c] += sh[c + s];
    __syncthreads();
  }
  float mean = sh[0] * (1.0f / kC);
  __syncthreads();
  float dv = v - mean;
  sh[c] = dv * dv;
  __syncthreads();
  for (int s = 64; s > 0; s >>= 1) {
    if (c < s) sh[c] += sh[c + s];
    __syncthreads();
  }
  float var = sh[0] * (1.0f / kC);
  float o = dv * rsqrtf(var + 1e-5f) * g[c] + b[c];
  src[idx] = o;
  sb[idx] = f2b(o);
  qb[idx] = f2b(o + pos[idx]);
}

// ---------------- pack merge GEMM A: A[p, n*128+c] = src[(n*4800+p)*128+c] ----------------
__global__ void merge_pack_kernel(const float* __restrict__ src, bf16_t* __restrict__ mA) {
  int t = blockIdx.x * blockDim.x + threadIdx.x;
  if (t >= kHW * (kNL * kC)) return;
  int k = t & 511;
  int p = t >> 9;
  int n = k >> 7;
  int c = k & 127;
  mA[t] = f2b(src[((size_t)(n * kHW + p)) * kC + c]);
}

// ---------------- bilinear 2x upsample, (h,w,c) layout ----------------
__global__ void resize_kernel(const float* __restrict__ m, float* __restrict__ r) {
  int t = blockIdx.x * blockDim.x + threadIdx.x;
  if (t >= kRH * kRW * kC) return;
  int c = t & 127;
  int pix = t >> 7;
  int ox = pix % kRW;
  int oy = pix / kRW;
  float sx = ((float)ox + 0.5f) * 0.5f - 0.5f;
  float sy = ((float)oy + 0.5f) * 0.5f - 0.5f;
  float fx0 = floorf(sx), fy0 = floorf(sy);
  int x0 = (int)fx0, y0 = (int)fy0;
  float fx = sx - fx0, fy = sy - fy0;
  int x0c = min(max(x0, 0), kW - 1), x1c = min(max(x0 + 1, 0), kW - 1);
  int y0c = min(max(y0, 0), kH - 1), y1c = min(max(y0 + 1, 0), kH - 1);
  float v00 = m[(size_t)(y0c * kW + x0c) * kC + c];
  float v01 = m[(size_t)(y0c * kW + x1c) * kC + c];
  float v10 = m[(size_t)(y1c * kW + x0c) * kC + c];
  float v11 = m[(size_t)(y1c * kW + x1c) * kC + c];
  r[t] = (1.0f - fy) * ((1.0f - fx) * v00 + fx * v01) + fy * ((1.0f - fx) * v10 + fx * v11);
}

// ---------------- 3x3 stride-1 conv + ReLU: (h,w,c) in -> NCHW out ----------------
__global__ void uconv_kernel(const float* __restrict__ r, const float* __restrict__ w,
                             const float* __restrict__ b, float* __restrict__ out) {
  __shared__ float patch[9 * kC];
  int pix = blockIdx.x;
  int oy = pix / kRW;
  int ox = pix % kRW;
  int t = threadIdx.x;
  for (int e = t; e < 9 * kC; e += 128) {
    int tap = e >> 7;
    int ci  = e & 127;
    int ky = tap / 3, kx = tap % 3;
    int iy = oy - 1 + ky, ix = ox - 1 + kx;
    float v = 0.0f;
    if (iy >= 0 && iy < kRH && ix >= 0 && ix < kRW)
      v = r[(size_t)(iy * kRW + ix) * kC + ci];   // coalesced across threads
    patch[e] = v;
  }
  __syncthreads();
  int o = t;
  float acc = b[o];
  const float* wo = w + (size_t)o * 1152;
  for (int ci = 0; ci < kC; ++ci) {
    float* pp = &patch[ci];
    const float* wp = wo + ci * 9;
#pragma unroll
    for (int tap = 0; tap < 9; ++tap)
      acc = fmaf(pp[tap * kC], wp[tap], acc);
  }
  out[((size_t)o * kRH + oy) * kRW + ox] = fmaxf(acc, 0.0f);
}

// ---------------- host-side GEMM dispatch ----------------
static void launch_gemm(int epi, const bf16_t* A, const bf16_t* B, const float* bias,
                        float* Cf, bf16_t* Cb, int M, int N, int K, hipStream_t s) {
  dim3 grid((unsigned)(M / 64), (unsigned)(N / 128)), blk(128);
  switch (epi) {
    case 0:  wmma_gemm_kernel<0><<<grid, blk, 0, s>>>(A, B, bias, Cf, Cb, M, N, K); break;
    case 1:  wmma_gemm_kernel<1><<<grid, blk, 0, s>>>(A, B, bias, Cf, Cb, M, N, K); break;
    default: wmma_gemm_kernel<2><<<grid, blk, 0, s>>>(A, B, bias, Cf, Cb, M, N, K); break;
  }
}

extern "C" void kernel_launch(void* const* d_in, const int* in_sizes, int n_in,
                              void* d_out, int out_size, void* d_ws, size_t ws_size,
                              hipStream_t stream) {
  (void)in_sizes; (void)n_in; (void)out_size; (void)ws_size;

  const float* x       = (const float*)d_in[0];
  const float* dconv_w = (const float*)d_in[1];
  const float* dconv_b = (const float*)d_in[2];
  const float* lvl_emb = (const float*)d_in[3];
  const float* W_off   = (const float*)d_in[4];
  const float* b_off   = (const float*)d_in[5];
  const float* W_attn  = (const float*)d_in[6];
  const float* b_attn  = (const float*)d_in[7];
  const float* W_val   = (const float*)d_in[8];
  const float* b_val   = (const float*)d_in[9];
  const float* W_out   = (const float*)d_in[10];
  const float* b_out   = (const float*)d_in[11];
  const float* ln1_g   = (const float*)d_in[12];
  const float* ln1_b   = (const float*)d_in[13];
  const float* W_ff1   = (const float*)d_in[14];
  const float* b_ff1   = (const float*)d_in[15];
  const float* W_ff2   = (const float*)d_in[16];
  const float* b_ff2   = (const float*)d_in[17];
  const float* ln2_g   = (const float*)d_in[18];
  const float* ln2_b   = (const float*)d_in[19];
  const float* merge_w = (const float*)d_in[20];
  const float* merge_b = (const float*)d_in[21];
  const float* uconv_w = (const float*)d_in[22];
  const float* uconv_b = (const float*)d_in[23];

  // ---- workspace carve (base assumed 256B aligned) ----
  char* base = (char*)d_ws;
  size_t off = 0;
  auto carve = [&](size_t bytes) -> void* {
    void* p = base + off;
    off += (bytes + 255) & ~(size_t)255;
    return p;
  };
  float*  src_f  = (float*)carve((size_t)kLq * kC * 4);
  float*  pos_f  = (float*)carve((size_t)kLq * kC * 4);
  float*  val_f  = (float*)carve((size_t)kLq * kC * 4);     // reused as merge output
  float*  y_f    = (float*)carve((size_t)kLq * kC * 4);     // reused as resized map
  float*  off_f  = (float*)carve((size_t)kLq * 256 * 4);    // reused as ff1 bf16
  float*  attn_f = (float*)carve((size_t)kLq * kC * 4);
  bf16_t* src_b  = (bf16_t*)carve((size_t)kLq * kC * 2);
  bf16_t* q_b    = (bf16_t*)carve((size_t)kLq * kC * 2);
  bf16_t* ao_b   = (bf16_t*)carve((size_t)kLq * kC * 2);    // reused as merge-A bf16
  size_t wbf_elems = (size_t)3 * 256 * 128 + (size_t)3 * 3 * 128 * 128 +
                     (size_t)3 * 512 * 128 + (size_t)3 * 128 * 512 + (size_t)128 * 512;
  bf16_t* wbf = (bf16_t*)carve(wbf_elems * 2);

  bf16_t* ff1_b = (bf16_t*)off_f;   // alias: off consumed before FF1 produced
  bf16_t* mA_b  = ao_b;             // alias: attn-out consumed before merge pack
  float*  m_f   = val_f;            // alias: value maps consumed before merge out
  float*  r_f   = y_f;              // alias: GEMM y consumed before resize

  bf16_t* w_off_b  = wbf;
  bf16_t* w_attn_b = w_off_b  + (size_t)3 * 256 * 128;
  bf16_t* w_val_b  = w_attn_b + (size_t)3 * 128 * 128;
  bf16_t* w_out_b  = w_val_b  + (size_t)3 * 128 * 128;
  bf16_t* w_ff1_b  = w_out_b  + (size_t)3 * 128 * 128;
  bf16_t* w_ff2_b  = w_ff1_b  + (size_t)3 * 512 * 128;
  bf16_t* w_mrg_b  = w_ff2_b  + (size_t)3 * 128 * 512;

  auto cvt = [&](const float* s, bf16_t* d, int n) {
    cvt_bf16_kernel<<<dim3((unsigned)((n + 255) / 256)), dim3(256), 0, stream>>>(s, d, n);
  };
  cvt(W_off,   w_off_b,  3 * 256 * 128);
  cvt(W_attn,  w_attn_b, 3 * 128 * 128);
  cvt(W_val,   w_val_b,  3 * 128 * 128);
  cvt(W_out,   w_out_b,  3 * 128 * 128);
  cvt(W_ff1,   w_ff1_b,  3 * 512 * 128);
  cvt(W_ff2,   w_ff2_b,  3 * 128 * 512);
  cvt(merge_w, w_mrg_b,  128 * 512);

  pos_kernel<<<dim3(9600), dim3(256), 0, stream>>>(lvl_emb, pos_f);
  dconv_kernel<<<dim3(kLq), dim3(128), 0, stream>>>(x, dconv_w, dconv_b, src_f);
  pack_q_kernel<<<dim3(9600), dim3(256), 0, stream>>>(src_f, pos_f, src_b, q_b);

  for (int i = 0; i < kL; ++i) {
    launch_gemm(0, src_b, w_val_b  + (size_t)i * 128 * 128, b_val  + i * 128, val_f,  nullptr, kLq, 128, 128, stream);
    launch_gemm(0, q_b,   w_off_b  + (size_t)i * 256 * 128, b_off  + i * 256, off_f,  nullptr, kLq, 256, 128, stream);
    launch_gemm(0, q_b,   w_attn_b + (size_t)i * 128 * 128, b_attn + i * 128, attn_f, nullptr, kLq, 128, 128, stream);
    softmax16_kernel<<<dim3((kLq * kNH + 255) / 256), dim3(256), 0, stream>>>(attn_f);
    deform_kernel<<<dim3(kLq), dim3(128), 0, stream>>>(val_f, off_f, attn_f, ao_b);
    launch_gemm(0, ao_b, w_out_b + (size_t)i * 128 * 128, b_out + i * 128, y_f, nullptr, kLq, 128, 128, stream);
    resln_kernel<<<dim3(kLq), dim3(128), 0, stream>>>(src_f, y_f, ln1_g + i * 128, ln1_b + i * 128,
                                                      pos_f, src_b, q_b);
    launch_gemm(1, src_b, w_ff1_b + (size_t)i * 512 * 128, b_ff1 + i * 512, nullptr, ff1_b, kLq, 512, 128, stream);
    launch_gemm(0, ff1_b, w_ff2_b + (size_t)i * 128 * 512, b_ff2 + i * 128, y_f, nullptr, kLq, 128, 512, stream);
    resln_kernel<<<dim3(kLq), dim3(128), 0, stream>>>(src_f, y_f, ln2_g + i * 128, ln2_b + i * 128,
                                                      pos_f, src_b, q_b);
  }

  merge_pack_kernel<<<dim3(9600), dim3(256), 0, stream>>>(src_f, mA_b);
  launch_gemm(2, mA_b, w_mrg_b, merge_b, m_f, nullptr, kHW, 128, 512, stream);
  resize_kernel<<<dim3(9600), dim3(256), 0, stream>>>(m_f, r_f);
  uconv_kernel<<<dim3(kRH * kRW), dim3(128), 0, stream>>>(r_f, uconv_w, uconv_b, (float*)d_out);
}
